// pLayer_85899346044
// MI455X (gfx1250) — matrix operand order repaired and checked by reading, not compiled, not run on hardware
//
#include <hip/hip_runtime.h>

#define NIN    1024
#define NOUT   512
#define NBATCH 512
#define NT     32
#define BTOT   (NBATCH * NT)   // 16384
#define GMINc  0.01f
#define GMAXc  10.0f
#define PGMINc 0.1f

typedef __attribute__((ext_vector_type(16))) _Float16 v16h;
typedef __attribute__((ext_vector_type(8)))  _Float16 v8h;
typedef __attribute__((ext_vector_type(8)))  float    v8f;

// ---- workspace layout (float offsets) ----
#define WS_APART 0      // 2 floats: A-term per prep block
#define WS_SX    1024   // 1024: sum_x per input row m
#define WS_SXX   2048   // 1024: sum_x^2 per input row m
#define WS_CZ    3072   // 512
#define WS_CU    3584   // 512
#define WS_GC    4096   // 512
#define WS_BPART 4608   // 1024: per-block sum z*u
#define WS_CPART 5632   // 1024: per-block sum z^2*gc
#define WS_WH    8192   // _Float16 region starts here: WhT[512*1024], then GhT

__device__ __forceinline__ float blockReduce256(float v, float* red) {
  const int tid = threadIdx.x;
  red[tid] = v; __syncthreads();
  #pragma unroll
  for (int s = 128; s > 0; s >>= 1) {
    if (tid < s) red[tid] += red[tid + s];
    __syncthreads();
  }
  const float r = red[0];
  __syncthreads();
  return r;
}

__device__ __forceinline__ v16h cat8(v8h lo, v8h hi) {
  return __builtin_shufflevector(lo, hi, 0,1,2,3,4,5,6,7,8,9,10,11,12,13,14,15);
}

// ---------------- per-row x statistics: Sx[m], Sxx[m] ----------------
__global__ void __launch_bounds__(256) k_stats(const float* __restrict__ x,
                                               float* __restrict__ sx,
                                               float* __restrict__ sxx) {
  __shared__ float red[256];
  const int m = blockIdx.x;
  const int tid = threadIdx.x;
  const int t = tid & 31, b0 = tid >> 5;
  float s = 0.f, s2 = 0.f;
  for (int b = b0; b < NBATCH; b += 8) {
    const float v = x[((size_t)b * NIN + m) * NT + t];
    s += v; s2 += v * v;
  }
  const float rs  = blockReduce256(s,  red);
  const float rs2 = blockReduce256(s2, red);
  if (tid == 0) { sx[m] = rs; sxx[m] = rs2; }
}

// ------- theta prep: effective weights (f16, transposed), constants -------
__global__ void __launch_bounds__(256) k_prep(const float* __restrict__ theta,
                                              const float* __restrict__ sx,
                                              const float* __restrict__ sxx,
                                              _Float16* __restrict__ WhT,
                                              _Float16* __restrict__ GhT,
                                              float* __restrict__ Cz,
                                              float* __restrict__ Cu,
                                              float* __restrict__ gc,
                                              float* __restrict__ apart) {
  __shared__ float red[256];
  const int n = blockIdx.x * blockDim.x + threadIdx.x;   // 0..511

  // pass 1: column sum of |theta_eff| and column min of |theta_raw|
  float S = 0.f, gmn = 3.4e38f;
  for (int m = 0; m < NIN + 2; ++m) {
    const float th  = theta[(size_t)m * NOUT + n];
    const float thc = fminf(fmaxf(th, -GMAXc), GMAXc);
    const float thq = (fabsf(thc) < GMINc) ? 0.f : thc;
    S   += fabsf(thq);
    gmn  = fminf(gmn, fabsf(th));
  }
  const float invS = 1.f / S;
  const float gs   = PGMINc / gmn;

  float czn = 0.f, cun = 0.f, gcn = 0.f, at = 0.f;
  for (int m = 0; m < NIN; ++m) {
    const float th  = theta[(size_t)m * NOUT + n];
    const float thc = fminf(fmaxf(th, -GMAXc), GMAXc);
    const float thq = (fabsf(thc) < GMINc) ? 0.f : thc;
    const bool  pos = (thq >= 0.f);
    const float ga  = fabsf(th) * gs;                    // g_tilde
    WhT[(size_t)n * NIN + m] = (_Float16)(thq * invS);   // W*pos - W*neg
    GhT[(size_t)n * NIN + m] = (_Float16)(pos ? ga : -ga);
    if (!pos) { czn += fabsf(thq) * invS; cun += ga; }
    gcn += ga;
    const float sxxm = sxx[m], sxm = sx[m];
    const float snn  = (float)BTOT - 2.f * sxm + sxxm;   // sum (1-x)^2
    at += ga * (pos ? sxxm : snn);
  }
  { // m = NIN: bias rail (x_ext = 1, x_neg = 0)
    const float th  = theta[(size_t)NIN * NOUT + n];
    const float thc = fminf(fmaxf(th, -GMAXc), GMAXc);
    const float thq = (fabsf(thc) < GMINc) ? 0.f : thc;
    const bool  pos = (thq >= 0.f);
    const float ga  = fabsf(th) * gs;
    if (pos) { czn += fabsf(thq) * invS; cun += ga; at += ga * (float)BTOT; }
    gcn += ga;
  }
  { // m = NIN+1: zero rail (x_ext = 0, x_neg = 0): only gc contribution
    const float th = theta[(size_t)(NIN + 1) * NOUT + n];
    gcn += fabsf(th) * gs;
  }
  Cz[n] = czn; Cu[n] = cun; gc[n] = gcn;
  const float atot = blockReduce256(at, red);
  if (threadIdx.x == 0) apart[blockIdx.x] = atot;
}

// ---------------- fused dual-GEMM + power + SG-MLP kernel ----------------
#define SXR 136   // lX row stride in halves (rows = k/m, cols = r), 128+8
#define SWR 40    // lW/lG row stride in halves (rows = n, cols = k/m), 32+8

__global__ void __launch_bounds__(256)
k_main(const float* __restrict__ x, const _Float16* __restrict__ WhT,
       const _Float16* __restrict__ GhT, const float* __restrict__ Cz,
       const float* __restrict__ Cu, const float* __restrict__ gc,
       const float* __restrict__ sw1, const float* __restrict__ sb1,
       const float* __restrict__ sw2, const float* __restrict__ sb2,
       float* __restrict__ out, float* __restrict__ bpart,
       float* __restrict__ cpart)
{
  __shared__ __align__(16) _Float16 lX[2][32 * SXR];
  __shared__ __align__(16) _Float16 lW[2][64 * SWR];
  __shared__ __align__(16) _Float16 lG[2][64 * SWR];
  __shared__ float red[256];

  const int tid  = threadIdx.x;
  const int lane = tid & 31;
  const int w    = tid >> 5;
  const int wr   = w & 3;            // r sub-block (32 rows each)
  const int wc   = w >> 2;           // n sub-block (32 cols each)
  const int nb     = blockIdx.x * 64;
  const int rowblk = blockIdx.y * 128;
  const int bbase  = blockIdx.y * 4; // 4 batch entries per 128-row block

  v8f cz[2][2] = {};
  v8f cu[2][2] = {};

  // X-tile loader indices: thread -> (k, 8-wide r group), 2 k's per thread
  const int xg = tid & 15;           // r-group: r = xg*8 (bl = xg>>2, t0 = (xg&3)*8)
  const int xk = tid >> 4;           // k in 0..15, also +16
  // W/G-tile loader indices: one 16B async copy per matrix per thread
  const int aml = (tid & 3) * 8;     // m offset within 32-wide K slab
  const int anl = tid >> 2;          // n row 0..63

  auto loadTileX = [&](int buf, int kt) {
    const int km = kt * 32;
    const int bl = xg >> 2;
    const int t0 = (xg & 3) * 8;
    #pragma unroll
    for (int q = 0; q < 2; ++q) {
      const int k = xk + q * 16;
      const float* src = &x[((size_t)(bbase + bl) * NIN + (km + k)) * NT + t0];
      const float4 f0 = *(const float4*)(src);
      const float4 f1 = *(const float4*)(src + 4);
      v8h h;
      h[0] = (_Float16)f0.x; h[1] = (_Float16)f0.y;
      h[2] = (_Float16)f0.z; h[3] = (_Float16)f0.w;
      h[4] = (_Float16)f1.x; h[5] = (_Float16)f1.y;
      h[6] = (_Float16)f1.z; h[7] = (_Float16)f1.w;
      *(v8h*)&lX[buf][k * SXR + xg * 8] = h;   // one ds_store_b128
    }
  };

  auto loadTilesWG = [&](int buf, int kt) {
    const int km = kt * 32;
    // byte offset into the f16 [n][m] arrays (fits 32 bits)
    const unsigned goff =
        (unsigned)((((unsigned)(nb + anl)) * NIN + (unsigned)(km + aml)) * 2u);
    const unsigned ldsW = (unsigned)(uintptr_t)&lW[buf][anl * SWR + aml];
    const unsigned ldsG = (unsigned)(uintptr_t)&lG[buf][anl * SWR + aml];
    // gfx1250 async DMA: 16B/lane global -> LDS, tracked by ASYNCcnt
    asm volatile("global_load_async_to_lds_b128 %0, %1, %2\n\t"
                 "global_load_async_to_lds_b128 %3, %4, %5"
                 :: "v"(ldsW), "v"(goff), "s"(WhT),
                    "v"(ldsG), "v"(goff), "s"(GhT)
                 : "memory");
  };

  loadTileX(0, 0);
  loadTilesWG(0, 0);
  asm volatile("s_wait_asynccnt 0x0" ::: "memory");
  __syncthreads();

  const int k0    = (lane >> 4) * 8;          // A-frag K chunk base per ISA layout
  const int nrow0 = wc * 32 + (lane & 15);    // A-frag row (M = n)
  const int r0    = wr * 32;

  for (int kt = 0; kt < 32; ++kt) {
    const int cur = kt & 1;
    if (kt + 1 < 32) {
      loadTileX(cur ^ 1, kt + 1);
      loadTilesWG(cur ^ 1, kt + 1);
    }

    v16h aw[2], ag[2], bx[2];
    #pragma unroll
    for (int ti = 0; ti < 2; ++ti) {
      const _Float16* wb = &lW[cur][(nrow0 + ti * 16) * SWR + k0];
      aw[ti] = cat8(*(const v8h*)wb, *(const v8h*)(wb + 16));
      const _Float16* gb = &lG[cur][(nrow0 + ti * 16) * SWR + k0];
      ag[ti] = cat8(*(const v8h*)gb, *(const v8h*)(gb + 16));
    }
    #pragma unroll
    for (int tj = 0; tj < 2; ++tj) {
      const _Float16* xb = &lX[cur][r0 + tj * 16 + (lane & 15)]; // B-frag: lane = N, elements = K (table order)
      #pragma unroll
      for (int s = 0; s < 16; ++s)
        bx[tj][s] = xb[(k0 + (s & 7) + ((s >> 3) << 4)) * SXR];
    }
    #pragma unroll
    for (int ti = 0; ti < 2; ++ti) {
      #pragma unroll
      for (int tj = 0; tj < 2; ++tj) {
        cz[ti][tj] = __builtin_amdgcn_wmma_f32_16x16x32_f16(
            false, aw[ti], false, bx[tj], (short)0, cz[ti][tj], false, false);
        cu[ti][tj] = __builtin_amdgcn_wmma_f32_16x16x32_f16(
            false, ag[ti], false, bx[tj], (short)0, cu[ti][tj], false, false);
      }
    }
    asm volatile("s_wait_asynccnt 0x0" ::: "memory");
    __syncthreads();
  }

  // ---- fused epilogue: constants, power partials, SG pointwise MLP ----
  float w1[8], b1[8], w2[8];
  #pragma unroll
  for (int j = 0; j < 8; ++j) { w1[j] = sw1[j]; b1[j] = sb1[j]; w2[j] = sw2[j]; }
  const float b2v = sb2[0];

  float bsum = 0.f, csum = 0.f;
  #pragma unroll
  for (int ti = 0; ti < 2; ++ti) {
    #pragma unroll
    for (int tj = 0; tj < 2; ++tj) {
      #pragma unroll
      for (int i = 0; i < 8; ++i) {
        const int n = nb + wc * 32 + ti * 16 + i + ((lane >> 4) << 3);
        const int r = rowblk + wr * 32 + tj * 16 + (lane & 15);
        const float zf = cz[ti][tj][i] + Cz[n];
        const float uf = cu[ti][tj][i] + Cu[n];
        bsum += zf * uf;
        csum += zf * zf * gc[n];
        float s = b2v;
        #pragma unroll
        for (int j = 0; j < 8; ++j) {
          const float e2 = __expf(-2.f * (zf * w1[j] + b1[j]));  // tanh via exp
          s += w2[j] * ((1.f - e2) / (1.f + e2));
        }
        const float aout = 1.f / (1.f + __expf(-s));
        out[(size_t)(r >> 5) * (NOUT * NT) + (size_t)n * NT + (r & 31)] = aout;
      }
    }
  }
  const float bs = blockReduce256(bsum, red);
  const float cs = blockReduce256(csum, red);
  if (tid == 0) {
    const int bid = blockIdx.y * gridDim.x + blockIdx.x;
    bpart[bid] = bs;
    cpart[bid] = cs;
  }
}

// ---------------- final deterministic combine ----------------
__global__ void __launch_bounds__(256) k_finish(const float* __restrict__ wsf,
                                                float* __restrict__ power) {
  __shared__ float red[256];
  const int tid = threadIdx.x;
  float b = 0.f, c = 0.f;
  for (int i = tid; i < 1024; i += 256) {
    b += wsf[WS_BPART + i];
    c += wsf[WS_CPART + i];
  }
  const float bs = blockReduce256(b, red);
  const float cs = blockReduce256(c, red);
  if (tid == 0) {
    const float A = wsf[WS_APART + 0] + wsf[WS_APART + 1];
    power[0] = (A - 2.f * bs + cs) * (1.f / (float)BTOT);
  }
}

extern "C" void kernel_launch(void* const* d_in, const int* in_sizes, int n_in,
                              void* d_out, int out_size, void* d_ws, size_t ws_size,
                              hipStream_t stream) {
  const float* x     = (const float*)d_in[0];
  const float* theta = (const float*)d_in[1];
  const float* sw1   = (const float*)d_in[2];
  const float* sb1   = (const float*)d_in[3];
  const float* sw2   = (const float*)d_in[4];
  const float* sb2   = (const float*)d_in[5];
  float* out = (float*)d_out;
  float* wsf = (float*)d_ws;

  _Float16* WhT = (_Float16*)(wsf + WS_WH);
  _Float16* GhT = WhT + (size_t)NIN * NOUT;

  hipLaunchKernelGGL(k_stats, dim3(NIN), dim3(256), 0, stream,
                     x, wsf + WS_SX, wsf + WS_SXX);
  hipLaunchKernelGGL(k_prep, dim3(2), dim3(256), 0, stream,
                     theta, wsf + WS_SX, wsf + WS_SXX, WhT, GhT,
                     wsf + WS_CZ, wsf + WS_CU, wsf + WS_GC, wsf + WS_APART);
  hipLaunchKernelGGL(k_main, dim3(NOUT / 64, BTOT / 128), dim3(256), 0, stream,
                     x, WhT, GhT, wsf + WS_CZ, wsf + WS_CU, wsf + WS_GC,
                     sw1, sb1, sw2, sb2, out, wsf + WS_BPART, wsf + WS_CPART);
  hipLaunchKernelGGL(k_finish, dim3(1), dim3(256), 0, stream,
                     wsf, out + (size_t)NBATCH * NOUT * NT);
}
